// CrossAttentionClassifier_67095979098632
// MI455X (gfx1250) — matrix operand order repaired and checked by reading, not compiled, not run on hardware
//
#include <hip/hip_runtime.h>

// ---------------------------------------------------------------------------
// CDNA5 (gfx1250) implementation: all GEMMs on v_wmma_f32_16x16x32_bf16.
// Main GEMM: 4 waves/block, 64x64 wave tile (4x4 WMMA accumulators),
// double-buffered LDS (one barrier per K-step), 1.0 WMMA:ds_load ratio.
// ---------------------------------------------------------------------------

typedef __attribute__((ext_vector_type(16))) __bf16 bf16x16;
typedef __attribute__((ext_vector_type(8)))  float  floatx8;

struct __align__(16) U4 { unsigned int x, y, z, w; };   // POD 16B vector

union FragAB { bf16x16 v; U4 q[2]; };

static __device__ __forceinline__ unsigned short f32_to_bf16(float f) {
    unsigned int u = __float_as_uint(f);
    u += 0x7fffu + ((u >> 16) & 1u);   // round-to-nearest-even
    return (unsigned short)(u >> 16);
}

static __device__ __forceinline__ float gelu_exact(float x) {
    return 0.5f * x * (1.0f + erff(x * 0.70710678118654752f));
}

static __device__ __forceinline__ floatx8 wmma_bf16(const FragAB& a, const FragAB& b, floatx8 c) {
    return __builtin_amdgcn_wmma_f32_16x16x32_bf16(
        false, a.v, false, b.v, (short)0, c, false, false);
}

// ---------------------------------------------------------------------------
// fp32 -> bf16 weight conversion (grid-stride)
// ---------------------------------------------------------------------------
__global__ void cvt_f32_bf16(const float* __restrict__ in,
                             unsigned short* __restrict__ out, long n) {
    long i = (long)blockIdx.x * blockDim.x + threadIdx.x;
    long stride = (long)gridDim.x * blockDim.x;
    for (; i < n; i += stride) out[i] = f32_to_bf16(in[i]);
}

// ---------------------------------------------------------------------------
// LayerNorm over `ncols` (256 thr/row), optional exact GELU, bf16 output
// ---------------------------------------------------------------------------
__global__ __launch_bounds__(256)
void ln_kernel(const float* __restrict__ x, const float* __restrict__ g,
               const float* __restrict__ b, unsigned short* __restrict__ out,
               int ncols, int act) {
    __shared__ float rs[256], rq[256];
    const int tid = threadIdx.x;
    const long row = blockIdx.x;
    const int cnt = ncols >> 8;            // 4 (E=1024) or 8 (2E)
    float v[8];
    float s = 0.f, s2 = 0.f;
    const float* xr = x + row * (long)ncols;
    for (int i = 0; i < cnt; ++i) { float t = xr[i * 256 + tid]; v[i] = t; s += t; s2 += t * t; }
    rs[tid] = s; rq[tid] = s2; __syncthreads();
    for (int off = 128; off; off >>= 1) {
        if (tid < off) { rs[tid] += rs[tid + off]; rq[tid] += rq[tid + off]; }
        __syncthreads();
    }
    const float inv_n = 1.0f / (float)ncols;
    const float mean = rs[0] * inv_n;
    const float var  = rq[0] * inv_n - mean * mean;
    const float rstd = rsqrtf(var + 1e-5f);
    unsigned short* orow = out + row * (long)ncols;
    for (int i = 0; i < cnt; ++i) {
        int c = i * 256 + tid;
        float y = (v[i] - mean) * rstd * g[c] + b[c];
        if (act) y = gelu_exact(y);
        orow[c] = f32_to_bf16(y);
    }
}

// LayerNorm over concat([xa,xb], dim=1) with total width 2048; bf16 out
__global__ __launch_bounds__(256)
void ln_concat_kernel(const float* __restrict__ xa, const float* __restrict__ xb,
                      const float* __restrict__ g, const float* __restrict__ b,
                      unsigned short* __restrict__ out) {
    __shared__ float rs[256], rq[256];
    const int tid = threadIdx.x;
    const long row = blockIdx.x;
    float v[8]; float s = 0.f, s2 = 0.f;
    const float* pa = xa + row * 1024;
    const float* pb = xb + row * 1024;
    for (int i = 0; i < 4; ++i) { float t = pa[i * 256 + tid]; v[i] = t;     s += t; s2 += t * t; }
    for (int i = 0; i < 4; ++i) { float t = pb[i * 256 + tid]; v[4 + i] = t; s += t; s2 += t * t; }
    rs[tid] = s; rq[tid] = s2; __syncthreads();
    for (int off = 128; off; off >>= 1) {
        if (tid < off) { rs[tid] += rs[tid + off]; rq[tid] += rq[tid + off]; }
        __syncthreads();
    }
    const float mean = rs[0] * (1.0f / 2048.0f);
    const float var  = rq[0] * (1.0f / 2048.0f) - mean * mean;
    const float rstd = rsqrtf(var + 1e-5f);
    unsigned short* orow = out + row * 2048;
    for (int i = 0; i < 4; ++i) {
        int c = i * 256 + tid;
        orow[c] = f32_to_bf16((v[i] - mean) * rstd * g[c] + b[c]);
    }
    for (int i = 0; i < 4; ++i) {
        int c = 1024 + i * 256 + tid;
        orow[c] = f32_to_bf16((v[4 + i] - mean) * rstd * g[c] + b[c]);
    }
}

// ---------------------------------------------------------------------------
// Main WMMA GEMM:  out = epilogue( A[M,K](bf16) @ W[N,K]^T(bf16) + bias )
// Block tile 128x128, BK=32, 4 waves (2x2), wave tile 64x64 (4x4 WMMA tiles).
// Double-buffered LDS, one barrier per K-step; next tile's global loads are
// issued before the WMMA burst so their latency hides under matrix compute.
// LDS row stride 40 halves (80B): 16B-aligned ds_load_b128, conflict-free.
// ACT: 0=none, 1=exact GELU.  HAS_RES: add f32 residual at same [row,col].
// ---------------------------------------------------------------------------
#define BM 128
#define BN 128
#define BK 32
#define LDT 40

template<int ACT, int HAS_RES>
__global__ __launch_bounds__(128)
void gemm_bf16_kernel(const unsigned short* __restrict__ A,
                      const unsigned short* __restrict__ W,
                      const float* __restrict__ bias,
                      const float* __restrict__ res,
                      float* __restrict__ out32,
                      unsigned short* __restrict__ out16,
                      int M, int N, int K) {
    __shared__ unsigned short sA[2][BM * LDT];
    __shared__ unsigned short sB[2][BN * LDT];

    const int tid  = threadIdx.x;
    const int wid  = tid >> 5;   // 0..3
    const int lane = tid & 31;
    const int l16  = lane & 15;
    const int lh   = lane >> 4;
    const int wm   = wid & 1;    // 0..1 -> 64-row slab
    const int wn   = wid >> 1;   // 0..1 -> 64-col slab
    const long m0 = (long)blockIdx.y * BM;
    const long n0 = (long)blockIdx.x * BN;

    // cooperative loaders: 1 thread per tile row, 4 x b128 per matrix
    const unsigned short* gA = A + (m0 + tid) * (long)K;
    const unsigned short* gB = W + (n0 + tid) * (long)K;

    floatx8 acc[4][4];
    const floatx8 vzero = {0.f, 0.f, 0.f, 0.f, 0.f, 0.f, 0.f, 0.f};
    #pragma unroll
    for (int i = 0; i < 4; ++i)
        #pragma unroll
        for (int j = 0; j < 4; ++j) acc[i][j] = vzero;

    U4 ra[4], rb[4];
    // preload K-step 0 into buffer 0
    #pragma unroll
    for (int s = 0; s < 4; ++s) {
        ra[s] = *reinterpret_cast<const U4*>(gA + s * 8);
        rb[s] = *reinterpret_cast<const U4*>(gB + s * 8);
    }
    #pragma unroll
    for (int s = 0; s < 4; ++s) {
        *reinterpret_cast<U4*>(&sA[0][tid * LDT + s * 8]) = ra[s];
        *reinterpret_cast<U4*>(&sB[0][tid * LDT + s * 8]) = rb[s];
    }
    __syncthreads();

    int buf = 0;
    for (int k0 = 0; k0 < K; k0 += BK) {
        const bool more = (k0 + BK) < K;
        if (more) {   // prefetch next K-step from global (latency hidden by WMMAs)
            #pragma unroll
            for (int s = 0; s < 4; ++s) {
                ra[s] = *reinterpret_cast<const U4*>(gA + k0 + BK + s * 8);
                rb[s] = *reinterpret_cast<const U4*>(gB + k0 + BK + s * 8);
            }
        }

        FragAB fa[4], fb[4];
        // A 16x32 bf16, contiguous k order chosen here (B matches): lanes 0-15 hold K 0..15, lanes 16-31 K 16..31
        #pragma unroll
        for (int mt = 0; mt < 4; ++mt) {
            const unsigned short* p = &sA[buf][(wm * 64 + mt * 16 + l16) * LDT + lh * 16];
            fa[mt].q[0] = *reinterpret_cast<const U4*>(p);
            fa[mt].q[1] = *reinterpret_cast<const U4*>(p + 8);
        }
        // B 32x16 bf16, same contiguous k order as A: lanes 0-15 hold K 0..15, lanes 16-31 K 16..31
        #pragma unroll
        for (int nt = 0; nt < 4; ++nt) {
            const unsigned short* p = &sB[buf][(wn * 64 + nt * 16 + l16) * LDT + lh * 16];
            fb[nt].q[0] = *reinterpret_cast<const U4*>(p);
            fb[nt].q[1] = *reinterpret_cast<const U4*>(p + 8);
        }
        #pragma unroll
        for (int mt = 0; mt < 4; ++mt)
            #pragma unroll
            for (int nt = 0; nt < 4; ++nt)
                acc[mt][nt] = wmma_bf16(fa[mt], fb[nt], acc[mt][nt]);

        if (more) {   // stage next tile into the other buffer
            #pragma unroll
            for (int s = 0; s < 4; ++s) {
                *reinterpret_cast<U4*>(&sA[buf ^ 1][tid * LDT + s * 8]) = ra[s];
                *reinterpret_cast<U4*>(&sB[buf ^ 1][tid * LDT + s * 8]) = rb[s];
            }
        }
        __syncthreads();
        buf ^= 1;
    }

    // epilogue: C/D layout -> row = r + 8*lh, col = l16 within each 16x16 tile
    #pragma unroll
    for (int mt = 0; mt < 4; ++mt) {
        #pragma unroll
        for (int nt = 0; nt < 4; ++nt) {
            const long col = n0 + wn * 64 + nt * 16 + l16;
            const float bcol = bias[col];
            #pragma unroll
            for (int r = 0; r < 8; ++r) {
                const long row = m0 + wm * 64 + mt * 16 + lh * 8 + r;
                const long idx = row * (long)N + col;
                float x = acc[mt][nt][r] + bcol;
                if (HAS_RES) x += res[idx];
                if (ACT == 1) x = gelu_exact(x);
                if (out32) out32[idx] = x;
                if (out16) out16[idx] = f32_to_bf16(x);
            }
        }
    }
}

// ---------------------------------------------------------------------------
// Classifier GEMM: N=32. 8 waves/block, each wave 16 rows x 32 cols (2 tiles).
// W (32xK bf16) streams from L2; no LDS needed.
// ---------------------------------------------------------------------------
__global__ __launch_bounds__(256)
void gemm_cls_kernel(const unsigned short* __restrict__ A,
                     const unsigned short* __restrict__ Wc,
                     const float* __restrict__ bias,
                     float* __restrict__ out, int K) {
    const int tid  = threadIdx.x;
    const int wid  = tid >> 5;
    const int lane = tid & 31;
    const int l16  = lane & 15;
    const int lh   = lane >> 4;
    const long m0 = (long)blockIdx.x * 128 + wid * 16;

    const floatx8 vzero = {0.f, 0.f, 0.f, 0.f, 0.f, 0.f, 0.f, 0.f};
    floatx8 acc[2]; acc[0] = vzero; acc[1] = vzero;

    const unsigned short* pa  = A  + (m0 + l16) * (long)K + lh * 16;
    const unsigned short* pb0 = Wc + (long)(l16)      * K + lh * 16;
    const unsigned short* pb1 = Wc + (long)(16 + l16) * K + lh * 16;

    for (int k = 0; k < K; k += 32) {
        FragAB fa, fb0, fb1;
        fa.q[0]  = *reinterpret_cast<const U4*>(pa  + k);
        fa.q[1]  = *reinterpret_cast<const U4*>(pa  + k + 8);
        fb0.q[0] = *reinterpret_cast<const U4*>(pb0 + k);
        fb0.q[1] = *reinterpret_cast<const U4*>(pb0 + k + 8);
        fb1.q[0] = *reinterpret_cast<const U4*>(pb1 + k);
        fb1.q[1] = *reinterpret_cast<const U4*>(pb1 + k + 8);
        acc[0] = wmma_bf16(fa, fb0, acc[0]);
        acc[1] = wmma_bf16(fa, fb1, acc[1]);
    }
    #pragma unroll
    for (int nt = 0; nt < 2; ++nt) {
        const int col = nt * 16 + l16;
        const float bcol = bias[col];
        #pragma unroll
        for (int r = 0; r < 8; ++r) {
            const long row = m0 + lh * 8 + r;
            out[row * 32 + col] = acc[nt][r] + bcol;
        }
    }
}

// ---------------------------------------------------------------------------
// Host-side orchestration
// ---------------------------------------------------------------------------
extern "C" void kernel_launch(void* const* d_in, const int* in_sizes, int n_in,
                              void* d_out, int out_size, void* d_ws, size_t ws_size,
                              hipStream_t stream) {
    (void)in_sizes; (void)n_in; (void)out_size; (void)ws_size;
    const int B_ = 16384, E_ = 1024, NC_ = 32, FF_ = 4096;
    typedef unsigned short u16;

    const float* lc_cls  = (const float*)d_in[0];
    const float* tab_emb = (const float*)d_in[1];
    const float* ln_lc_g = (const float*)d_in[2];
    const float* ln_lc_b = (const float*)d_in[3];
    const float* ln_tb_g = (const float*)d_in[4];
    const float* ln_tb_b = (const float*)d_in[5];
    // alc: Wq=6 Wk=7 Wv=8 bq=9 bk=10 bv=11 Wo=12 bo=13  (Wq/Wk/bq/bk dead)
    const float* alc_Wv = (const float*)d_in[8];
    const float* alc_bv = (const float*)d_in[11];
    const float* alc_Wo = (const float*)d_in[12];
    const float* alc_bo = (const float*)d_in[13];
    const float* atb_Wv = (const float*)d_in[16];
    const float* atb_bv = (const float*)d_in[19];
    const float* atb_Wo = (const float*)d_in[20];
    const float* atb_bo = (const float*)d_in[21];
    const float* flc_W1 = (const float*)d_in[22];
    const float* flc_b1 = (const float*)d_in[23];
    const float* flc_W2 = (const float*)d_in[24];
    const float* flc_b2 = (const float*)d_in[25];
    const float* ftb_W1 = (const float*)d_in[26];
    const float* ftb_b1 = (const float*)d_in[27];
    const float* ftb_W2 = (const float*)d_in[28];
    const float* ftb_b2 = (const float*)d_in[29];
    const float* on_g   = (const float*)d_in[30];
    const float* on_b   = (const float*)d_in[31];
    const float* c_W1   = (const float*)d_in[32];
    const float* c_b1   = (const float*)d_in[33];
    const float* c_ln_g = (const float*)d_in[34];
    const float* c_ln_b = (const float*)d_in[35];
    const float* c_W2   = (const float*)d_in[36];
    const float* c_b2   = (const float*)d_in[37];
    float* out = (float*)d_out;

    // workspace carve-up
    char* ws = (char*)d_ws;
    size_t off = 0;
    auto carve = [&](size_t bytes) -> void* {
        void* p = ws + off; off += (bytes + 255) & ~(size_t)255; return p;
    };
    const size_t BE2 = (size_t)B_ * E_ * 2, BE4 = (size_t)B_ * E_ * 4;
    u16*   S0 = (u16*)carve(BE2);                 // lc_n -> lc_cross(bf16) -> cls_in
    u16*   S1 = (u16*)carve(BE2);                 // tab_n -> tab_cross(bf16)
    u16*   S2 = (u16*)carve(BE2);                 // v_lc
    u16*   S3 = (u16*)carve(BE2);                 // v_tab
    float* S4 = (float*)carve(BE4);               // lc_cross f32 -> lc_out -> c1
    float* S5 = (float*)carve(BE4);               // tab_cross f32 -> tab_out
    u16*   S6 = (u16*)carve((size_t)B_ * FF_ * 2);// h_lc / h_tab / fused
    u16* wAlcV = (u16*)carve((size_t)E_ * E_ * 2);
    u16* wAlcO = (u16*)carve((size_t)E_ * E_ * 2);
    u16* wAtbV = (u16*)carve((size_t)E_ * E_ * 2);
    u16* wAtbO = (u16*)carve((size_t)E_ * E_ * 2);
    u16* wF1L  = (u16*)carve((size_t)FF_ * E_ * 2);
    u16* wF2L  = (u16*)carve((size_t)E_ * FF_ * 2);
    u16* wF1T  = (u16*)carve((size_t)FF_ * E_ * 2);
    u16* wF2T  = (u16*)carve((size_t)E_ * FF_ * 2);
    u16* wC1   = (u16*)carve((size_t)E_ * 2 * E_ * 2);
    u16* wC2   = (u16*)carve((size_t)NC_ * E_ * 2);

    // --- weight conversion to bf16 ---
    const long EE = (long)E_ * E_, FE = (long)FF_ * E_;
    cvt_f32_bf16<<<2048, 256, 0, stream>>>(alc_Wv, wAlcV, EE);
    cvt_f32_bf16<<<2048, 256, 0, stream>>>(alc_Wo, wAlcO, EE);
    cvt_f32_bf16<<<2048, 256, 0, stream>>>(atb_Wv, wAtbV, EE);
    cvt_f32_bf16<<<2048, 256, 0, stream>>>(atb_Wo, wAtbO, EE);
    cvt_f32_bf16<<<4096, 256, 0, stream>>>(flc_W1, wF1L, FE);
    cvt_f32_bf16<<<4096, 256, 0, stream>>>(flc_W2, wF2L, FE);
    cvt_f32_bf16<<<4096, 256, 0, stream>>>(ftb_W1, wF1T, FE);
    cvt_f32_bf16<<<4096, 256, 0, stream>>>(ftb_W2, wF2T, FE);
    cvt_f32_bf16<<<2048, 256, 0, stream>>>(c_W1, wC1, 2 * EE);
    cvt_f32_bf16<<<256,  256, 0, stream>>>(c_W2, wC2, (long)NC_ * E_);

    // --- pre-attention layernorms ---
    ln_kernel<<<B_, 256, 0, stream>>>(lc_cls,  ln_lc_g, ln_lc_b, S0, E_, 0);
    ln_kernel<<<B_, 256, 0, stream>>>(tab_emb, ln_tb_g, ln_tb_b, S1, E_, 0);

    const dim3 gEE(E_ / BN, B_ / BM);    // N=1024
    const dim3 gEF(FF_ / BN, B_ / BM);   // N=4096

    // --- attention (softmax over 1 key == identity): V then O projections ---
    gemm_bf16_kernel<0,0><<<gEE, 128, 0, stream>>>(S1, wAlcV, alc_bv, nullptr, nullptr, S2, B_, E_, E_);
    gemm_bf16_kernel<0,0><<<gEE, 128, 0, stream>>>(S0, wAtbV, atb_bv, nullptr, nullptr, S3, B_, E_, E_);
    gemm_bf16_kernel<0,1><<<gEE, 128, 0, stream>>>(S2, wAlcO, alc_bo, lc_cls,  S4, S0, B_, E_, E_);
    gemm_bf16_kernel<0,1><<<gEE, 128, 0, stream>>>(S3, wAtbO, atb_bo, tab_emb, S5, S1, B_, E_, E_);

    // --- per-side FFN with residual ---
    gemm_bf16_kernel<1,0><<<gEF, 128, 0, stream>>>(S0, wF1L, flc_b1, nullptr, nullptr, S6, B_, FF_, E_);
    gemm_bf16_kernel<0,1><<<gEE, 128, 0, stream>>>(S6, wF2L, flc_b2, S4, S4, nullptr, B_, E_, FF_);
    gemm_bf16_kernel<1,0><<<gEF, 128, 0, stream>>>(S1, wF1T, ftb_b1, nullptr, nullptr, S6, B_, FF_, E_);
    gemm_bf16_kernel<0,1><<<gEE, 128, 0, stream>>>(S6, wF2T, ftb_b2, S5, S5, nullptr, B_, E_, FF_);

    // --- fused concat LN (width 2048) ---
    ln_concat_kernel<<<B_, 256, 0, stream>>>(S4, S5, on_g, on_b, S6);

    // --- classifier trunk: c_W1 (K=2048) -> LN -> GELU -> c_W2 (N=32) ---
    gemm_bf16_kernel<0,0><<<gEE, 128, 0, stream>>>(S6, wC1, c_b1, nullptr, S4, nullptr, B_, E_, 2 * E_);
    ln_kernel<<<B_, 256, 0, stream>>>(S4, c_ln_g, c_ln_b, S0, E_, 1);
    gemm_cls_kernel<<<B_ / 128, 256, 0, stream>>>(S0, wC2, c_b2, out, E_);
}
